// PairRelationEncoder_62775241998442
// MI455X (gfx1250) — matrix operand order repaired and checked by reading, not compiled, not run on hardware
//
#include <hip/hip_runtime.h>

// Problem constants (from reference): B=4, N=4096, D=64, E=131072, R=64
#define B_ 4
#define N_ 4096
#define D_ 64
#define E_ 131072
#define R_ 64

#define ROWS_PER_TILE 128
#define NTILES ((B_ * E_) / ROWS_PER_TILE)   // 4096
#define NBLK   512                            // persistent blocks; 8 tiles each

typedef __attribute__((ext_vector_type(16))) __bf16 v16bf;
typedef __attribute__((ext_vector_type(8)))  float  v8f;

union FragU {
  unsigned u[8];
  uint4    q[2];
  v16bf    v;
};

// Pack two f32 into packed bf16 (round-half-up) with a single v_perm_b32.
__device__ __forceinline__ unsigned pack2(float lo, float hi) {
  unsigned ul = __builtin_bit_cast(unsigned, lo) + 0x8000u;
  unsigned uh = __builtin_bit_cast(unsigned, hi) + 0x8000u;
  // result bytes: [3]=uh.b3 [2]=uh.b2 [1]=ul.b3 [0]=ul.b2
  return __builtin_amdgcn_perm(uh, ul, 0x07060302u);
}
__device__ __forceinline__ unsigned short f2bf(float f) {
  return (unsigned short)((__builtin_bit_cast(unsigned, f) + 0x8000u) >> 16);
}

__global__ __launch_bounds__(256)
void pair_encoder_kernel(const float* __restrict__ z,
                         const int*   __restrict__ pairs,
                         const float* __restrict__ W1,
                         const float* __restrict__ b1,
                         const float* __restrict__ W2,
                         const float* __restrict__ b2,
                         float*       __restrict__ out)
{
  // Padded LDS (bank-conflict-free strides): total 61440 B < 64 KB
  __shared__ __align__(16) unsigned short sW1[64][264];  // W1 bf16, rows padded 256->264
  __shared__ __align__(16) unsigned short sW2[64][72];   // W2 bf16, rows padded 64->72
  __shared__ __align__(16) unsigned short sH [128][72];  // relu(h) bf16 staging (per-wave rows)

  const int tid  = threadIdx.x;
  const int lane = tid & 31;
  const int wave = tid >> 5;
  const int half = lane >> 4;     // 0 or 1 (lane half)
  const int nrow = lane & 15;

  // ---------------- Stage weights f32 -> bf16 into LDS (once per block) ----------------
  {
    const int row = tid >> 2;              // 0..63
    const int c0  = (tid & 3) * 64;        // quarter of a W1 row
    const float4* s1 = (const float4*)(W1 + row * 256 + c0);
    #pragma unroll
    for (int i = 0; i < 16; ++i) {
      float4 v = s1[i];
      uint2 p; p.x = pack2(v.x, v.y); p.y = pack2(v.z, v.w);
      *(uint2*)&sW1[row][c0 + i * 4] = p;
    }
    const int c2 = (tid & 3) * 16;         // quarter of a W2 row
    const float4* s2 = (const float4*)(W2 + row * 64 + c2);
    #pragma unroll
    for (int i = 0; i < 4; ++i) {
      float4 v = s2[i];
      uint2 p; p.x = pack2(v.x, v.y); p.y = pack2(v.z, v.w);
      *(uint2*)&sW2[row][c2 + i * 4] = p;
    }
  }

  // Bias fragments (constant across tiles): lane column = t*16 + nrow
  float bias1[4], bias2[4];
  #pragma unroll
  for (int t = 0; t < 4; ++t) {
    bias1[t] = b1[t * 16 + nrow];
    bias2[t] = b2[t * 16 + nrow];
  }

  __syncthreads();   // weights staged

  const int o     = half * 8;    // A-fragment per-lane K offset (16-bit A layout)
  const int bko   = half * 8;    // B-fragment per-lane K offset
  const int mbase = half * 8;    // C/D layout: lane<16 -> M 0..7, lane>=16 -> M 8..15
  const int hrow0 = wave * 16;

  // ------- Hoist ALL weight B-fragments into registers (loop-invariant) -------
  // GEMM1: 8 K-chunks x 4 N-tiles = 32 frags (256 VGPRs)
  // GEMM2: 2 K-chunks x 4 N-tiles =  8 frags ( 64 VGPRs)
  FragU bW1[8][4];
  #pragma unroll
  for (int kc = 0; kc < 8; ++kc) {
    const int k0 = kc * 32 + bko;
    #pragma unroll
    for (int t = 0; t < 4; ++t) {
      bW1[kc][t].q[0] = *(const uint4*)&sW1[t * 16 + nrow][k0];
      bW1[kc][t].q[1] = *(const uint4*)&sW1[t * 16 + nrow][k0 + 16];
    }
  }
  FragU bW2[2][4];
  #pragma unroll
  for (int cc = 0; cc < 2; ++cc) {
    const int kb = cc * 32 + bko;
    #pragma unroll
    for (int t = 0; t < 4; ++t) {
      bW2[cc][t].q[0] = *(const uint4*)&sW2[t * 16 + nrow][kb];
      bW2[cc][t].q[1] = *(const uint4*)&sW2[t * 16 + nrow][kb + 16];
    }
  }

  #pragma unroll 1
  for (int tile = blockIdx.x; tile < NTILES; tile += NBLK) {
    // ---------------- Gather z rows; features live in registers ----------------
    const long long g = (long long)tile * ROWS_PER_TILE + hrow0 + nrow;
    const int b  = (int)(g >> 17);           // E = 2^17
    const int e  = (int)(g & (E_ - 1));
    const int pi = pairs[2 * e + 0];
    const int pj = pairs[2 * e + 1];
    const float* zi = z + ((long long)b * N_ + pi) * D_;
    const float* zj = z + ((long long)b * N_ + pj) * D_;

    // lane<16 holds K {0..7,16..23} of each 32-chunk, lane>=16 holds {8..15,24..31}
    float ai[4][8], aj[4][8];                // 4 groups of 8 consecutive d-values
    #pragma unroll
    for (int grp = 0; grp < 4; ++grp) {
      const int d = o + grp * 16;
      float4 x0 = *(const float4*)(zi + d);
      float4 x1 = *(const float4*)(zi + d + 4);
      float4 y0 = *(const float4*)(zj + d);
      float4 y1 = *(const float4*)(zj + d + 4);
      ai[grp][0] = x0.x; ai[grp][1] = x0.y; ai[grp][2] = x0.z; ai[grp][3] = x0.w;
      ai[grp][4] = x1.x; ai[grp][5] = x1.y; ai[grp][6] = x1.z; ai[grp][7] = x1.w;
      aj[grp][0] = y0.x; aj[grp][1] = y0.y; aj[grp][2] = y0.z; aj[grp][3] = y0.w;
      aj[grp][4] = y1.x; aj[grp][5] = y1.y; aj[grp][6] = y1.z; aj[grp][7] = y1.w;
    }

    // ---------------- GEMM1: h = pair(16x256) @ W1^T(256x64) ----------------
    v8f acc[4];
    {
      v8f zero = {0.f, 0.f, 0.f, 0.f, 0.f, 0.f, 0.f, 0.f};
      #pragma unroll
      for (int t = 0; t < 4; ++t) acc[t] = zero;
    }

    #pragma unroll
    for (int f = 0; f < 4; ++f) {            // feature group: zi, zj, zi-zj, zi*zj
      #pragma unroll
      for (int cc = 0; cc < 2; ++cc) {       // two K=32 chunks per 64-feature group
        const int g0 = 2 * cc, g1 = 2 * cc + 1;
        FragU af;
        #pragma unroll
        for (int i = 0; i < 4; ++i) {
          float a0 = ai[g0][2*i],   a1 = ai[g0][2*i+1];
          float c0 = aj[g0][2*i],   c1 = aj[g0][2*i+1];
          float v0 = (f == 0) ? a0 : (f == 1) ? c0 : (f == 2) ? (a0 - c0) : (a0 * c0);
          float v1 = (f == 0) ? a1 : (f == 1) ? c1 : (f == 2) ? (a1 - c1) : (a1 * c1);
          af.u[i] = pack2(v0, v1);
        }
        #pragma unroll
        for (int i = 0; i < 4; ++i) {
          float a0 = ai[g1][2*i],   a1 = ai[g1][2*i+1];
          float c0 = aj[g1][2*i],   c1 = aj[g1][2*i+1];
          float v0 = (f == 0) ? a0 : (f == 1) ? c0 : (f == 2) ? (a0 - c0) : (a0 * c0);
          float v1 = (f == 0) ? a1 : (f == 1) ? c1 : (f == 2) ? (a1 - c1) : (a1 * c1);
          af.u[4 + i] = pack2(v0, v1);
        }

        const int kc = f * 2 + cc;
        #pragma unroll
        for (int t = 0; t < 4; ++t) {        // 4 N-tiles of 16
          acc[t] = __builtin_amdgcn_wmma_f32_16x16x32_bf16(
              false, af.v, false, bW1[kc][t].v, (short)0, acc[t], false, false);
        }
      }
    }

    // ------- bias + ReLU, stage h (bf16) via per-wave LDS transpose (no barrier) -------
    #pragma unroll
    for (int t = 0; t < 4; ++t) {
      #pragma unroll
      for (int j = 0; j < 8; ++j) {
        float v = acc[t][j] + bias1[t];
        v = v > 0.f ? v : 0.f;
        sH[hrow0 + mbase + j][t * 16 + nrow] = f2bf(v);
      }
    }

    // ---------------- GEMM2: out = relu(h)(16x64) @ W2^T(64x64) ----------------
    v8f acc2[4];
    {
      v8f zero = {0.f, 0.f, 0.f, 0.f, 0.f, 0.f, 0.f, 0.f};
      #pragma unroll
      for (int t = 0; t < 4; ++t) acc2[t] = zero;
    }

    const int hr = hrow0 + nrow;
    #pragma unroll
    for (int cc = 0; cc < 2; ++cc) {
      FragU af;
      const int ka = cc * 32 + o;            // A-fragment pattern on h rows
      af.q[0] = *(const uint4*)&sH[hr][ka];
      af.q[1] = *(const uint4*)&sH[hr][ka + 16];
      #pragma unroll
      for (int t = 0; t < 4; ++t) {
        acc2[t] = __builtin_amdgcn_wmma_f32_16x16x32_bf16(
            false, af.v, false, bW2[cc][t].v, (short)0, acc2[t], false, false);
      }
    }

    // ---------------- bias + streaming f32 store ----------------
    const long long rowbase = (long long)tile * ROWS_PER_TILE + hrow0;
    #pragma unroll
    for (int t = 0; t < 4; ++t) {
      const int col = t * 16 + nrow;
      #pragma unroll
      for (int j = 0; j < 8; ++j) {
        long long r = rowbase + mbase + j;
        __builtin_nontemporal_store(acc2[t][j] + bias2[t], out + r * 64 + col);
      }
    }
  }
}

extern "C" void kernel_launch(void* const* d_in, const int* in_sizes, int n_in,
                              void* d_out, int out_size, void* d_ws, size_t ws_size,
                              hipStream_t stream) {
  const float* z     = (const float*)d_in[0];
  const int*   pairs = (const int*)  d_in[1];   // harness delivers integer inputs as int32
  const float* W1    = (const float*)d_in[2];
  const float* b1    = (const float*)d_in[3];
  const float* W2    = (const float*)d_in[4];
  const float* b2    = (const float*)d_in[5];
  float* out = (float*)d_out;

  dim3 grid(NBLK);     // persistent blocks, each loops over NTILES/NBLK = 8 tiles
  dim3 block(256);     // 8 wave32
  pair_encoder_kernel<<<grid, block, 0, stream>>>(z, pairs, W1, b1, W2, b2, out);
}
